// EGATConv28_83330955477489
// MI455X (gfx1250) — matrix operand order repaired and checked by reading, not compiled, not run on hardware
//
#include <hip/hip_runtime.h>
#include <hip/hip_bf16.h>
#include <math.h>

// ---------------- Problem constants ----------------
#define NN   20000
#define EE   320000
#define DEG  16          // EE / NN
#define HH   4
#define DIN  128
#define DOUT 128
#define KFE  384         // 2*DOUT + DIN
#define NFN  512         // H*DOUT
#define MT   4           // M-tiles (of 16 rows) per workgroup in edge GEMMs
#define APAD 392         // padded LDS row stride for K=384 panel (bf16 elems)
#define EPAD 136         // padded LDS row stride for K=128 panel (bf16 elems)

typedef __bf16 bf16_t;
typedef __attribute__((ext_vector_type(16))) __bf16 v16bf;
typedef __attribute__((ext_vector_type(8)))  __bf16 v8bf;
typedef __attribute__((ext_vector_type(8)))  float  v8f;

// ---------------- WMMA helpers ----------------
__device__ __forceinline__ v8f wmma_bf16(v16bf a, v16bf b, v8f c) {
  // D = A(16x32 bf16) * B(32x16 bf16) + C(16x16 f32)
  return __builtin_amdgcn_wmma_f32_16x16x32_bf16(false, a, false, b, (short)0, c,
                                                 false, false);
}

// A fragment from fp32 row-major source. p = &row[kchunk*32 + half*8].
__device__ __forceinline__ v16bf a_from_f32(const float* __restrict__ p) {
  v16bf r;
#pragma unroll
  for (int i = 0; i < 8; ++i) {
    r[i]     = (bf16_t)p[i];
    r[i + 8] = (bf16_t)p[i + 16];
  }
  return r;
}

// A fragment from bf16 row-major source (global or LDS): 8 bf16 at p, 8 at p+16.
__device__ __forceinline__ v16bf a_from_bf16(const bf16_t* p) {
  v8bf lo = *(const v8bf*)p;
  v8bf hi = *(const v8bf*)(p + 16);
  v16bf r;
#pragma unroll
  for (int i = 0; i < 8; ++i) { r[i] = lo[i]; r[i + 8] = hi[i]; }
  return r;
}

// B fragment from transposed (K-contiguous) bf16 weights: 8 bf16 at p, 8 at p+16.
__device__ __forceinline__ v16bf b_from_bf16(const bf16_t* __restrict__ p) {
  v8bf lo = *(const v8bf*)p;
  v8bf hi = *(const v8bf*)(p + 16);
  v16bf r;
#pragma unroll
  for (int i = 0; i < 8; ++i) { r[i] = lo[i]; r[i + 8] = hi[i]; }
  return r;
}

// ---------------- Kernel 0: zero a scratch region ----------------
__global__ void k_zero(float* __restrict__ p, int n) {
  int i = blockIdx.x * blockDim.x + threadIdx.x;
  if (i < n) p[i] = 0.0f;
}

// ---------------- Kernel 1: per-edge scalar MLPs ----------------
__global__ void k_edge_scalar(const float* __restrict__ ew_in, const int* __restrict__ etype,
    const float* __restrict__ c0w, const float* __restrict__ c0b,
    const float* __restrict__ c2w, const float* __restrict__ c2b,
    const float* __restrict__ v0w, const float* __restrict__ v0b,
    const float* __restrict__ v2w, const float* __restrict__ v2b,
    const float* __restrict__ v4w, const float* __restrict__ v4b,
    const float* __restrict__ mean_scale, const float* __restrict__ var_scale,
    const float* __restrict__ lambda_raw,
    float* __restrict__ wm_out, float* __restrict__ wv_out,
    float* __restrict__ eff_out, float* __restrict__ lam_out) {
  int e = blockIdx.x * blockDim.x + threadIdx.x;
  if (e >= EE) return;
  float ew = ew_in[e];
  int   et = etype[e];
  // c-MLP: 1 -> 8 -> 1, tanh
  float hs = c2b[0];
#pragma unroll
  for (int j = 0; j < 8; ++j) hs += fmaxf(ew * c0w[j] + c0b[j], 0.0f) * c2w[j];
  float delta = tanhf(hs);
  float sp_m  = log1pf(expf(mean_scale[et]));
  float wm    = fminf(fmaxf(ew + delta, 0.0f), 1.0f) * sp_m;
  // v-MLP: 1 -> 16 -> 8 -> 1
  float h16[16];
#pragma unroll
  for (int j = 0; j < 16; ++j) h16[j] = fmaxf(ew * v0w[j] + v0b[j], 0.0f);
  float s4 = v4b[0];
#pragma unroll
  for (int k = 0; k < 8; ++k) {
    float s = v2b[k];
#pragma unroll
    for (int j = 0; j < 16; ++j) s += h16[j] * v2w[j * 8 + k];
    s4 += fmaxf(s, 0.0f) * v4w[k];
  }
  float sig  = 1.0f / (1.0f + expf(-s4));
  float sp_v = log1pf(expf(var_scale[et]));
  float wv   = sig * sp_v + 1e-6f;
  float eff  = fminf(fmaxf(wm / (1.0f + wv), 0.001f), 1.0f);
  float sp_l = log1pf(expf(lambda_raw[et]));
  wm_out[e]  = wm;
  wv_out[e]  = wv;
  eff_out[e] = eff;
  lam_out[e] = sp_l * eff;
}

// ---------------- Kernel 2: weight transpose + bf16 convert ----------------
// src: [K, Nc] row-major fp32 -> dst: [Nc, K] bf16
__global__ void k_transpose_bf16(const float* __restrict__ src, bf16_t* __restrict__ dst,
                                 int K, int Nc) {
  int i = blockIdx.x * blockDim.x + threadIdx.x;
  if (i >= K * Nc) return;
  int n = i / K, k = i % K;
  dst[i] = (bf16_t)src[(size_t)k * Nc + n];
}

// ---------------- Kernel 3: h = nfeats @ W_fn + b_fn  (WMMA) ----------------
__global__ void __launch_bounds__(256)
k_h_gemm(const float* __restrict__ nfeats, const bf16_t* __restrict__ WfnT,
         const float* __restrict__ b_fn, bf16_t* __restrict__ h_bf) {
  int wave = threadIdx.x >> 5;
  int lane = threadIdx.x & 31;
  int half = lane >> 4, l16 = lane & 15;
  int m0 = blockIdx.x * 16;
  const float* arow = nfeats + (size_t)(m0 + l16) * DIN;
  v16bf afrag[4];
#pragma unroll
  for (int kc = 0; kc < 4; ++kc) afrag[kc] = a_from_f32(arow + kc * 32 + half * 8);
  for (int nt = wave; nt < NFN / 16; nt += 8) {
    int n0 = nt * 16;
    int col = n0 + l16;
    v8f c = {};
#pragma unroll
    for (int kc = 0; kc < 4; ++kc) {
      v16bf b = b_from_bf16(WfnT + (size_t)col * DIN + kc * 32 + half * 8);
      c = wmma_bf16(afrag[kc], b, c);
    }
    float bias = b_fn[col];
#pragma unroll
    for (int v = 0; v < 8; ++v) {
      int row = m0 + half * 8 + v;
      h_bf[(size_t)row * NFN + col] = (bf16_t)(c[v] + bias);
    }
  }
}

// ---------------- Kernel 4: hm = head-mean of h ----------------
__global__ void k_hm(const bf16_t* __restrict__ h_bf, bf16_t* __restrict__ hm_bf) {
  int i = blockIdx.x * blockDim.x + threadIdx.x;
  if (i >= NN * DIN) return;
  int n = i >> 7, d = i & 127;
  const bf16_t* r = h_bf + (size_t)n * NFN;
  float s = ((float)r[d] + (float)r[d + 128] + (float)r[d + 256] + (float)r[d + 384]) * 0.25f;
  hm_bf[i] = (bf16_t)s;
}

// ---------------- Kernel 5: ef / ef_b edge GEMMs (WMMA, LDS A, 4 M-tiles) ----------------
__global__ void __launch_bounds__(256)
k_ef_gemm(const float* __restrict__ efeats, const bf16_t* __restrict__ WepT,
          const bf16_t* __restrict__ WepbT, const float* __restrict__ b_ep,
          const float* __restrict__ b_epb, const float* __restrict__ W_epb_full,
          const float* __restrict__ eff_w,
          bf16_t* __restrict__ ef_bf, bf16_t* __restrict__ efb_bf) {
  __shared__ __align__(16) bf16_t Atile[64 * EPAD];
  int t   = threadIdx.x;
  int m00 = blockIdx.x * (16 * MT);      // 64 edges per block
  {
    // stage A panel (64 x 128) fp32 -> bf16 into LDS
    int r = t >> 2, p = t & 3;           // thread -> (row, quarter of 32 elems)
    const float* sp = efeats + (size_t)(m00 + r) * DIN + p * 32;
    bf16_t* dp = Atile + r * EPAD + p * 32;
#pragma unroll
    for (int i = 0; i < 32; ++i) dp[i] = (bf16_t)sp[i];
  }
  __syncthreads();
  int wave = t >> 5, lane = t & 31, half = lane >> 4, l16 = lane & 15;
  int n0  = wave * 16;                   // 8 waves cover Nout = 128
  int col = n0 + l16;
  v8f c0[MT] = {};
  v8f c1[MT] = {};
#pragma unroll
  for (int kc = 0; kc < 4; ++kc) {
    v16bf b0 = b_from_bf16(WepT  + (size_t)col * DIN + kc * 32 + half * 8);
    v16bf b1 = b_from_bf16(WepbT + (size_t)col * DIN + kc * 32 + half * 8);
#pragma unroll
    for (int mt = 0; mt < MT; ++mt) {
      v16bf a = a_from_bf16(Atile + (mt * 16 + l16) * EPAD + kc * 32 + half * 8);
      c0[mt] = wmma_bf16(a, b0, c0[mt]);
      c1[mt] = wmma_bf16(a, b1, c1[mt]);
    }
  }
  float be    = b_ep[col];
  float beb   = b_epb[col];
  float wlast = W_epb_full[128 * DIN + col];   // row 128 of W_epb (eff_w weight)
#pragma unroll
  for (int mt = 0; mt < MT; ++mt) {
#pragma unroll
    for (int v = 0; v < 8; ++v) {
      int row = m00 + mt * 16 + half * 8 + v;
      ef_bf[(size_t)row * DIN + col]  = (bf16_t)(c0[mt][v] + be);
      efb_bf[(size_t)row * DIN + col] = (bf16_t)(c1[mt][v] + beb + eff_w[row] * wlast);
    }
  }
}

// ---------------- Kernel 6: f_out = leaky(stack @ W_fe) + attn logits ----------------
// 64 edges per block; A panel [64 x 384] bf16 staged in LDS; B reused across 4 M-tiles.
__global__ void __launch_bounds__(256)
k_fout_gemm(const bf16_t* __restrict__ hm_bf, const bf16_t* __restrict__ ef_bf,
            const bf16_t* __restrict__ WfeT, const float* __restrict__ W_attn,
            const int* __restrict__ src, const int* __restrict__ dst,
            float* __restrict__ fout, float* __restrict__ a_out) {
  __shared__ __align__(16) bf16_t Atile[64 * APAD];
  int t   = threadIdx.x;
  int m00 = blockIdx.x * (16 * MT);      // 64 edges per block
  if (t < 192) {
    // thread -> (row r, segment s): copy 128 bf16 (256 B)
    int r = t & 63, s = t >> 6;
    int e = m00 + r;
    const bf16_t* sp;
    if (s == 0)      sp = hm_bf + (size_t)src[e] * DIN;   // hm[src]
    else if (s == 1) sp = ef_bf + (size_t)e * DIN;        // ef
    else             sp = hm_bf + (size_t)dst[e] * DIN;   // hm[dst]
    bf16_t* dp = Atile + r * APAD + s * DIN;
#pragma unroll
    for (int i = 0; i < 16; ++i)
      *(v8bf*)(dp + i * 8) = *(const v8bf*)(sp + i * 8);
  }
  __syncthreads();
  int wave = t >> 5, lane = t & 31, half = lane >> 4, l16 = lane & 15;
  for (int nt = wave; nt < NFN / 16; nt += 8) {
    int n0  = nt * 16;
    int col = n0 + l16;
    v8f acc[MT] = {};
    const bf16_t* bcol = WfeT + (size_t)col * KFE;
#pragma unroll
    for (int kc = 0; kc < 12; ++kc) {
      v16bf b = b_from_bf16(bcol + kc * 32 + half * 8);
#pragma unroll
      for (int mt = 0; mt < MT; ++mt) {
        v16bf a = a_from_bf16(Atile + (mt * 16 + l16) * APAD + kc * 32 + half * 8);
        acc[mt] = wmma_bf16(a, b, acc[mt]);
      }
    }
    int   hh  = n0 >> 7;              // head index, constant within a 16-col tile
    float wat = W_attn[col & 127];
#pragma unroll
    for (int mt = 0; mt < MT; ++mt) {
      float part[8];
#pragma unroll
      for (int v = 0; v < 8; ++v) {
        int row = m00 + mt * 16 + half * 8 + v;
        float x = acc[mt][v];
        x = x > 0.0f ? x : 0.01f * x;            // leaky_relu(0.01)
        fout[(size_t)row * NFN + col] = x;
        part[v] = x * wat;
      }
      // reduce across the 16 lanes of this half (same rows, different cols)
#pragma unroll
      for (int m = 1; m < 16; m <<= 1) {
#pragma unroll
        for (int v = 0; v < 8; ++v) part[v] += __shfl_xor(part[v], m, 32);
      }
      if (l16 == 0) {
#pragma unroll
        for (int v = 0; v < 8; ++v)
          atomicAdd(&a_out[(size_t)(m00 + mt * 16 + half * 8 + v) * HH + hh], part[v]);
      }
    }
  }
}

// ---------------- Kernel 7: per-node softmax + aggregation ----------------
// dst = repeat(arange(N), 16): node n owns edges [16n, 16n+16)
__global__ void __launch_bounds__(128)
k_node(const float* __restrict__ a, const float* __restrict__ lam_w,
       const bf16_t* __restrict__ h_bf, const bf16_t* __restrict__ efb_bf,
       const float* __restrict__ nfeats, const int* __restrict__ src,
       float* __restrict__ hout, float* __restrict__ agg, bf16_t* __restrict__ agg_bf) {
  int n = blockIdx.x;
  int t = threadIdx.x;
  __shared__ float alpha_s[DEG][HH];
  __shared__ float a_s[DEG][HH];
  __shared__ float lam_s[DEG];
  __shared__ int   src_s[DEG];
  __shared__ float laminv_s;
  size_t e0 = (size_t)n * DEG;
  if (t < DEG * HH) a_s[t >> 2][t & 3] = a[e0 * HH + t];
  if (t >= 64 && t < 64 + DEG) {
    lam_s[t - 64] = lam_w[e0 + (t - 64)];
    src_s[t - 64] = src[e0 + (t - 64)];
  }
  __syncthreads();
  if (t < HH) {
    float mx = -3.4e38f;
#pragma unroll
    for (int e = 0; e < DEG; ++e) mx = fmaxf(mx, a_s[e][t]);
    float sum = 0.0f;
#pragma unroll
    for (int e = 0; e < DEG; ++e) { float ex = expf(a_s[e][t] - mx); alpha_s[e][t] = ex; sum += ex; }
    float inv = 1.0f / sum;
#pragma unroll
    for (int e = 0; e < DEG; ++e) alpha_s[e][t] *= inv;
  }
  if (t == HH) {
    float s = 0.0f;
#pragma unroll
    for (int e = 0; e < DEG; ++e) s += lam_s[e];
    laminv_s = 1.0f / (s + 1e-8f);
  }
  __syncthreads();
  // thread t covers output channels t, t+128, t+256, t+384 (head 0..3)
  float acc0 = 0.f, acc1 = 0.f, acc2 = 0.f, acc3 = 0.f, accA = 0.f;
#pragma unroll 4
  for (int e = 0; e < DEG; ++e) {
    int sr = src_s[e];
    const bf16_t* hr = h_bf + (size_t)sr * NFN;
    acc0 += alpha_s[e][0] * (float)hr[t];
    acc1 += alpha_s[e][1] * (float)hr[t + 128];
    acc2 += alpha_s[e][2] * (float)hr[t + 256];
    acc3 += alpha_s[e][3] * (float)hr[t + 384];
    accA += lam_s[e] * nfeats[(size_t)sr * DIN + t] * (float)efb_bf[(e0 + e) * DIN + t];
  }
  size_t ob = (size_t)n * NFN;
  hout[ob + t]       = acc0;
  hout[ob + 128 + t] = acc1;
  hout[ob + 256 + t] = acc2;
  hout[ob + 384 + t] = acc3;
  float ag = accA * laminv_s;
  agg[(size_t)n * DIN + t]    = ag;
  agg_bf[(size_t)n * DIN + t] = (bf16_t)ag;
}

// ---------------- Kernel 8: h_out += broadcast(agg @ W_Bm + b_Bm)  (WMMA) ----------------
__global__ void __launch_bounds__(256)
k_agg_gemm(const bf16_t* __restrict__ agg_bf, const bf16_t* __restrict__ WBmT,
           const float* __restrict__ b_Bm, float* __restrict__ hout) {
  int wave = threadIdx.x >> 5;
  int lane = threadIdx.x & 31;
  int half = lane >> 4, l16 = lane & 15;
  int m0 = blockIdx.x * 16;
  const bf16_t* arow = agg_bf + (size_t)(m0 + l16) * DIN;
  v16bf afrag[4];
#pragma unroll
  for (int kc = 0; kc < 4; ++kc) afrag[kc] = a_from_bf16(arow + kc * 32 + half * 8);
  int n0  = wave * 16;            // 8 waves cover Nout = 128
  int col = n0 + l16;
  v8f c = {};
#pragma unroll
  for (int kc = 0; kc < 4; ++kc) {
    v16bf b = b_from_bf16(WBmT + (size_t)col * DIN + kc * 32 + half * 8);
    c = wmma_bf16(afrag[kc], b, c);
  }
  float bias = b_Bm[col];
#pragma unroll
  for (int v = 0; v < 8; ++v) {
    int row = m0 + half * 8 + v;
    float val = c[v] + bias;
    size_t ob = (size_t)row * NFN + col;
    hout[ob]       += val;
    hout[ob + 128] += val;
    hout[ob + 256] += val;
    hout[ob + 384] += val;
  }
}

// ---------------- Host launch ----------------
extern "C" void kernel_launch(void* const* d_in, const int* in_sizes, int n_in,
                              void* d_out, int out_size, void* d_ws, size_t ws_size,
                              hipStream_t stream) {
  const float* nfeats      = (const float*)d_in[0];
  const float* efeats      = (const float*)d_in[1];
  const float* edge_weight = (const float*)d_in[2];
  const float* W_ep        = (const float*)d_in[3];
  const float* b_ep        = (const float*)d_in[4];
  const float* W_epb       = (const float*)d_in[5];
  const float* b_epb       = (const float*)d_in[6];
  const float* W_fn        = (const float*)d_in[7];
  const float* b_fn        = (const float*)d_in[8];
  const float* W_fe        = (const float*)d_in[9];
  const float* W_attn      = (const float*)d_in[10];
  const float* W_Bm        = (const float*)d_in[11];
  const float* b_Bm        = (const float*)d_in[12];
  const float* lambda_raw  = (const float*)d_in[13];
  const float* mean_scale  = (const float*)d_in[14];
  const float* var_scale   = (const float*)d_in[15];
  const float* c0_w        = (const float*)d_in[16];
  const float* c0_b        = (const float*)d_in[17];
  const float* c2_w        = (const float*)d_in[18];
  const float* c2_b        = (const float*)d_in[19];
  const float* v0_w        = (const float*)d_in[20];
  const float* v0_b        = (const float*)d_in[21];
  const float* v2_w        = (const float*)d_in[22];
  const float* v2_b        = (const float*)d_in[23];
  const float* v4_w        = (const float*)d_in[24];
  const float* v4_b        = (const float*)d_in[25];
  const int*   src         = (const int*)d_in[26];
  const int*   dst         = (const int*)d_in[27];
  const int*   edge_type   = (const int*)d_in[28];

  // Output layout (tuple concat, fp32): h_out, f_out, wm, wv
  float* out   = (float*)d_out;
  float* hout  = out;                                  // N*H*DOUT
  float* fout  = out + (size_t)NN * NFN;               // E*H*DE
  float* wm_o  = out + (size_t)NN * NFN + (size_t)EE * NFN;
  float* wv_o  = wm_o + EE;

  // Workspace layout
  char*  ws  = (char*)d_ws;
  size_t off = 0;
  auto take = [&](size_t bytes) { char* p = ws + off; off += (bytes + 255) & ~(size_t)255; return p; };
  float*  ws_eff    = (float*)take((size_t)EE * 4);
  float*  ws_lam    = (float*)take((size_t)EE * 4);
  float*  ws_a      = (float*)take((size_t)EE * HH * 4);
  float*  ws_agg    = (float*)take((size_t)NN * DIN * 4);
  bf16_t* ws_hm     = (bf16_t*)take((size_t)NN * DIN * 2);
  bf16_t* ws_h      = (bf16_t*)take((size_t)NN * NFN * 2);
  bf16_t* ws_ef     = (bf16_t*)take((size_t)EE * DIN * 2);
  bf16_t* ws_efb    = (bf16_t*)take((size_t)EE * DIN * 2);
  bf16_t* ws_agg_bf = (bf16_t*)take((size_t)NN * DIN * 2);
  bf16_t* ws_WfnT   = (bf16_t*)take((size_t)NFN * DIN * 2);
  bf16_t* ws_WfeT   = (bf16_t*)take((size_t)NFN * KFE * 2);
  bf16_t* ws_WepT   = (bf16_t*)take((size_t)DIN * DIN * 2);
  bf16_t* ws_WepbT  = (bf16_t*)take((size_t)DIN * DIN * 2);
  bf16_t* ws_WBmT   = (bf16_t*)take((size_t)DIN * DIN * 2);

  // 0) zero attention-logit accumulator (atomics target)
  k_zero<<<(EE * HH + 255) / 256, 256, 0, stream>>>(ws_a, EE * HH);

  // 1) per-edge scalar MLPs -> wm, wv (d_out) + eff_w, lam_w (ws)
  k_edge_scalar<<<EE / 256, 256, 0, stream>>>(edge_weight, edge_type,
      c0_w, c0_b, c2_w, c2_b, v0_w, v0_b, v2_w, v2_b, v4_w, v4_b,
      mean_scale, var_scale, lambda_raw, wm_o, wv_o, ws_eff, ws_lam);

  // 2) weight transpose+convert to bf16 [Nout, K]
  k_transpose_bf16<<<(NFN * DIN + 255) / 256, 256, 0, stream>>>(W_fn, ws_WfnT, DIN, NFN);
  k_transpose_bf16<<<(NFN * KFE + 255) / 256, 256, 0, stream>>>(W_fe, ws_WfeT, KFE, NFN);
  k_transpose_bf16<<<(DIN * DIN + 255) / 256, 256, 0, stream>>>(W_ep, ws_WepT, DIN, DIN);
  k_transpose_bf16<<<(DIN * DIN + 255) / 256, 256, 0, stream>>>(W_epb, ws_WepbT, DIN, DIN);
  k_transpose_bf16<<<(DIN * DIN + 255) / 256, 256, 0, stream>>>(W_Bm, ws_WBmT, DIN, DIN);

  // 3) h = nfeats @ W_fn + b_fn   (WMMA bf16)
  k_h_gemm<<<NN / 16, 256, 0, stream>>>(nfeats, ws_WfnT, b_fn, ws_h);

  // 4) hm = head-mean(h)
  k_hm<<<(NN * DIN + 255) / 256, 256, 0, stream>>>(ws_h, ws_hm);

  // 5) ef / ef_b edge GEMMs   (WMMA bf16, LDS-staged A, 64 edges/block)
  k_ef_gemm<<<EE / (16 * MT), 256, 0, stream>>>(efeats, ws_WepT, ws_WepbT, b_ep, b_epb,
                                                W_epb, ws_eff, ws_ef, ws_efb);

  // 6) f_out = leaky([hm[src]|ef|hm[dst]] @ W_fe) + attention logits
  //    (WMMA bf16, LDS-staged A panel, B reused across 4 M-tiles)
  k_fout_gemm<<<EE / (16 * MT), 256, 0, stream>>>(ws_hm, ws_ef, ws_WfeT, W_attn,
                                                  src, dst, fout, ws_a);

  // 7) per-node softmax + h_topo + agg (dst is sorted: 16 edges per node)
  k_node<<<NN, 128, 0, stream>>>(ws_a, ws_lam, ws_h, ws_efb, nfeats, src,
                                 hout, ws_agg, ws_agg_bf);

  // 8) h_out += broadcast(agg @ W_Bm + b_Bm)   (WMMA bf16)
  k_agg_gemm<<<NN / 16, 256, 0, stream>>>(ws_agg_bf, ws_WBmT, b_Bm, hout);

  (void)in_sizes; (void)n_in; (void)out_size; (void)ws_size;
}
